// Dice_77515569758537
// MI455X (gfx1250) — 8-bit matrix operand order repaired and checked by reading, compiled, not run on hardware
//
#include <hip/hip_runtime.h>
#include <hip/hip_bf16.h>

typedef __attribute__((ext_vector_type(4))) float v4f;
typedef __attribute__((ext_vector_type(4))) int   v4i;
typedef __attribute__((ext_vector_type(8))) int   v8i;

#define NBATCH 2
#define NCLS   8
#define SVOX   (128LL * 128LL * 128LL)
#define TILE   128
#define TILES_PER_WAVE  8
#define WAVES_PER_BLOCK 8
#define THREADS (WAVES_PER_BLOCK * 32)
#define VOX_PER_BLOCK (WAVES_PER_BLOCK * TILES_PER_WAVE * TILE)
#define BLOCKS_PER_BATCH ((int)(SVOX / VOX_PER_BLOCK))

__device__ __forceinline__ unsigned byte_eq01(unsigned x, unsigned clsSplat) {
  unsigned v = x ^ clsSplat;
  unsigned z = v | (v >> 1) | (v >> 2) | (v >> 3);
  return ~z & 0x01010101u;
}

__global__ __launch_bounds__(THREADS)
void dice_count_kernel(const float* __restrict__ pred,
                       const int*   __restrict__ ref,
                       int*         __restrict__ gcnt) {
  __shared__ unsigned sPred[WAVES_PER_BLOCK][32];
  __shared__ unsigned sRef [WAVES_PER_BLOCK][32];
  __shared__ int lcnt[24];

  const int tid  = threadIdx.x;
  const int w    = tid >> 5;
  const int lane = tid & 31;
  const int b    = blockIdx.x / BLOCKS_PER_BATCH;
  const int blk  = blockIdx.x % BLOCKS_PER_BATCH;

  if (tid < 24) lcnt[tid] = 0;
  __syncthreads();

  const float* predB = pred + (size_t)b * NCLS * SVOX;
  const int*   refB  = ref  + (size_t)b * SVOX;

  const int      cls      = lane & 15;
  const unsigned clsSplat = (unsigned)cls * 0x01010101u;
  const unsigned is8      = 0u - (unsigned)(cls == 8);
  const int      hi       = lane >> 4;
  const int      hiA2     = hi * 2;
  const int      hiB4     = hi * 4;

  v8i acc = {};

  const size_t waveBase =
      ((size_t)blk * WAVES_PER_BLOCK + w) * (size_t)(TILES_PER_WAVE * TILE);

  for (int t = 0; t < TILES_PER_WAVE; ++t) {
    const size_t v0 = waveBase + (size_t)t * TILE + (size_t)lane * 4;

    v4f bv = *(const v4f*)(predB + v0);
    unsigned l0 = 0, l1 = 0, l2 = 0, l3 = 0;
#pragma unroll
    for (int c = 1; c < NCLS; ++c) {
      v4f x = *(const v4f*)(predB + (size_t)c * SVOX + v0);
      bool g0 = x.x > bv.x; bv.x = g0 ? x.x : bv.x; l0 = g0 ? (unsigned)c : l0;
      bool g1 = x.y > bv.y; bv.y = g1 ? x.y : bv.y; l1 = g1 ? (unsigned)c : l1;
      bool g2 = x.z > bv.z; bv.z = g2 ? x.z : bv.z; l2 = g2 ? (unsigned)c : l2;
      bool g3 = x.w > bv.w; bv.w = g3 ? x.w : bv.w; l3 = g3 ? (unsigned)c : l3;
    }
    unsigned pPack = l0 | (l1 << 8) | (l2 << 16) | (l3 << 24);

    v4i r = *(const v4i*)(refB + v0);
    unsigned rPack = ((unsigned)r.x & 7u)         | (((unsigned)r.y & 7u) << 8) |
                     (((unsigned)r.z & 7u) << 16) | (((unsigned)r.w & 7u) << 24);

    sPred[w][lane] = pPack;
    sRef [w][lane] = rPack;
    asm volatile("s_wait_dscnt 0" ::: "memory");

#if defined(__gfx1250__)
#pragma unroll
    for (int kt = 0; kt < 2; ++kt) {
      const int tw = kt * 16;
      v8i Amat, Bmat;
#pragma unroll
      for (int j = 0; j < 8; ++j) {
        unsigned av = sPred[w][tw + ((j >> 1) << 2) + (j & 1) + hiA2];
        unsigned ea = byte_eq01(av, clsSplat) | (is8 & 0x01010101u);
        Amat[j] = (int)ea;
        unsigned bw = sRef[w][tw + ((j >> 1) << 2) + (j & 1) + hiA2];
        unsigned eb = byte_eq01(bw, clsSplat) | (is8 & 0x01010101u);
        Bmat[j] = (int)eb;
      }
      acc = __builtin_amdgcn_wmma_i32_16x16x64_iu8(
           false, Amat,  false, Bmat, acc,
           false,  false);
    }
#endif
    asm volatile("" ::: "memory");
  }

#pragma unroll
  for (int j = 0; j < 8; ++j) {
    int v = acc[j];
    if (lane == j) atomicAdd(&lcnt[j], v);
    if (lane == 8) atomicAdd(&lcnt[8 + j], v);
  }
  if (lane >= 16 && lane < 24)
    atomicAdd(&lcnt[16 + (lane - 16)], acc[0]);
  __syncthreads();

  if (tid < 24) atomicAdd(&gcnt[b * 24 + tid], lcnt[tid]);
}

__global__ void zero_cnt_kernel(int* __restrict__ g) {
  if (threadIdx.x < NBATCH * 24) g[threadIdx.x] = 0;
}

__global__ void finalize_kernel(const int* __restrict__ g,
                                float* __restrict__ out) {
  if (threadIdx.x == 0 && blockIdx.x == 0) {
    float mean = 0.f;
#pragma unroll
    for (int b = 0; b < NBATCH; ++b) {
      float sd = 0.f, sw = 0.f;
#pragma unroll
      for (int c = 1; c < NCLS; ++c) {
        float inter = (float)g[b * 24 + c];
        float ps    = (float)g[b * 24 + 8 + c];
        float rs    = (float)g[b * 24 + 16 + c];
        if (rs > 0.f) {
          float u = ps + rs;
          sd += 2.f * inter / (u > 0.f ? u : 1.f);
          sw += 1.f;
        }
      }
      mean += sd / sw;
    }
    out[0] = mean / (float)NBATCH;
  }
}

extern "C" void kernel_launch(void* const* d_in, const int* in_sizes, int n_in,
                              void* d_out, int out_size, void* d_ws, size_t ws_size,
                              hipStream_t stream) {
  (void)in_sizes; (void)n_in; (void)out_size; (void)ws_size;
  const float* pred = (const float*)d_in[0];
  const int*   ref  = (const int*)d_in[1];
  float*       out  = (float*)d_out;
  int*         cnt  = (int*)d_ws;

  zero_cnt_kernel<<<1, 64, 0, stream>>>(cnt);
  dice_count_kernel<<<NBATCH * BLOCKS_PER_BATCH, THREADS, 0, stream>>>(pred, ref, cnt);
  finalize_kernel<<<1, 32, 0, stream>>>(cnt, out);
}
